// ExpertsLayer_34119220199660
// MI455X (gfx1250) — matrix operand order repaired and checked by reading, not compiled, not run on hardware
//
#include <hip/hip_runtime.h>
#include <hip/hip_bf16.h>

typedef __bf16 bf16_t;
typedef __attribute__((ext_vector_type(16))) __bf16 v16bf;
typedef __attribute__((ext_vector_type(8)))  float  v8f;

#define TOKENS 8192
#define DMODEL 1024
#define NEXPERT 8
#define DFFN 2048
#define NSLOTS (TOKENS * 2)   /* top_k = 2 */

#define TM 64
#define TN 64
#define TKK 32

// ---- gfx1250 async global->LDS path (ASYNCcnt), with safe fallback ----
#if defined(__HIP_DEVICE_COMPILE__) && \
    __has_builtin(__builtin_amdgcn_global_load_async_to_lds_b64) && \
    __has_builtin(__builtin_amdgcn_s_wait_asynccnt)
#define ASYNC_LDS 1
#else
#define ASYNC_LDS 0
#endif

#if ASYNC_LDS
typedef int v2i_t __attribute__((ext_vector_type(2)));
typedef __attribute__((address_space(1))) v2i_t gas_v2i;
typedef __attribute__((address_space(3))) v2i_t las_v2i;
__device__ __forceinline__ void async_copy_b64(void* lds_dst, const void* gsrc) {
  __builtin_amdgcn_global_load_async_to_lds_b64(
      (gas_v2i*)(void*)gsrc, (las_v2i*)lds_dst, 0, 0);
}
__device__ __forceinline__ void wait_async_all() {
  __builtin_amdgcn_s_wait_asynccnt(0);
}
#endif

__device__ __forceinline__ float sigmoidf_(float x) {
  return 1.0f / (1.0f + __expf(-x));
}

// ---------------- f32 -> bf16 conversion (4-wide) ----------------
__global__ void cvt_f32_bf16(const float* __restrict__ s, bf16_t* __restrict__ d, int n) {
  int i = (blockIdx.x * blockDim.x + threadIdx.x) * 4;
  if (i + 3 < n) {
    float4 v = *(const float4*)(s + i);
    bf16_t o[4] = { (bf16_t)v.x, (bf16_t)v.y, (bf16_t)v.z, (bf16_t)v.w };
    *(uint2*)(d + i) = *(const uint2*)o;
  } else {
    for (int j = 0; j < 4 && i + j < n; ++j) d[i + j] = (bf16_t)s[i + j];
  }
}

__global__ void zero_f32(float* __restrict__ p, int n) {
  int i = blockIdx.x * blockDim.x + threadIdx.x;
  if (i < n) p[i] = 0.0f;
}

// ---------------- Router: logits, softmax, top-2, shared gate ----------------
__global__ void router_kernel(const float* __restrict__ x,      // [T][D]
                              const float* __restrict__ gw,     // [D][E]
                              const float* __restrict__ segw,   // [D][1]
                              float* __restrict__ gate_out,     // [T][E] (d_out tail)
                              int*   __restrict__ topi,         // [T][2]
                              float* __restrict__ topw,         // [T][2]
                              float* __restrict__ sgate,        // [T]
                              int*   __restrict__ counts)       // [E]
{
  int t = blockIdx.x * blockDim.x + threadIdx.x;
  if (t >= TOKENS) return;
  const float* xr = x + (size_t)t * DMODEL;
  float acc[NEXPERT];
#pragma unroll
  for (int e = 0; e < NEXPERT; ++e) acc[e] = 0.0f;
  float sg = 0.0f;
  for (int d = 0; d < DMODEL; ++d) {
    float xv = xr[d];
    const float* g = gw + (size_t)d * NEXPERT;
#pragma unroll
    for (int e = 0; e < NEXPERT; ++e) acc[e] += xv * g[e];
    sg += xv * segw[d];
  }
#pragma unroll
  for (int e = 0; e < NEXPERT; ++e) gate_out[(size_t)t * NEXPERT + e] = acc[e];
  // softmax (fp32)
  float mx = acc[0];
#pragma unroll
  for (int e = 1; e < NEXPERT; ++e) mx = fmaxf(mx, acc[e]);
  float p[NEXPERT], den = 0.0f;
#pragma unroll
  for (int e = 0; e < NEXPERT; ++e) { p[e] = __expf(acc[e] - mx); den += p[e]; }
  float inv = 1.0f / den;
#pragma unroll
  for (int e = 0; e < NEXPERT; ++e) p[e] *= inv;
  // top-2 (earliest index wins ties, matching lax.top_k)
  int i0 = 0;
#pragma unroll
  for (int e = 1; e < NEXPERT; ++e) if (p[e] > p[i0]) i0 = e;
  int i1 = (i0 == 0) ? 1 : 0;
#pragma unroll
  for (int e = 0; e < NEXPERT; ++e) if (e != i0 && p[e] > p[i1]) i1 = e;
  topi[t * 2 + 0] = i0; topw[t * 2 + 0] = p[i0];
  topi[t * 2 + 1] = i1; topw[t * 2 + 1] = p[i1];
  atomicAdd(&counts[i0], 1);
  atomicAdd(&counts[i1], 1);
  sgate[t] = sigmoidf_(sg);
}

__global__ void prefix_kernel(const int* __restrict__ counts,
                              int* __restrict__ slotBase, int* __restrict__ fill) {
  if (threadIdx.x == 0 && blockIdx.x == 0) {
    int s = 0;
    for (int e = 0; e < NEXPERT; ++e) { slotBase[e] = s; s += counts[e]; fill[e] = 0; }
    slotBase[NEXPERT] = s;
  }
}

__global__ void build_lists(const int* __restrict__ topi, const float* __restrict__ topw,
                            const int* __restrict__ slotBase, int* __restrict__ fill,
                            int* __restrict__ tokList, float* __restrict__ tokW) {
  int t = blockIdx.x * blockDim.x + threadIdx.x;
  if (t >= TOKENS) return;
#pragma unroll
  for (int k = 0; k < 2; ++k) {
    int e = topi[t * 2 + k];
    int pos = atomicAdd(&fill[e], 1);
    int s = slotBase[e] + pos;
    tokList[s] = t;
    tokW[s] = topw[t * 2 + k];
  }
}

// ---------------- WMMA fragment helpers (gfx1250 bf16 16x16x32 layouts) ----------------
union FragA { uint4 q[2]; v16bf v; };

__device__ __forceinline__ v16bf load_a_frag(const bf16_t* rowBase, int half) {
  // A 16x32 bf16: element i -> K = (i/8)*16 + half*8 + i%8  => two contiguous 16B chunks.
  FragA f;
  f.q[0] = *(const uint4*)(rowBase + half * 8);
  f.q[1] = *(const uint4*)(rowBase + 16 + half * 8);
  return f.v;
}

__device__ __forceinline__ v16bf load_b_frag(const bf16_t* btRow, int half) {
  // B staged transposed in LDS: Bt[n][k]; element i -> K = (i/8)*16 + half*8 + i%8 (two 16B chunks).
  FragA f;
  f.q[0] = *(const uint4*)(btRow + half * 8);
  f.q[1] = *(const uint4*)(btRow + 16 + half * 8);
  return f.v;
}

// ---------------- Fused gate/up GEMM + SiLU*up -> H (bf16) ----------------
__launch_bounds__(256)
__global__ void moe_gateup(const bf16_t* __restrict__ X,    // [T][D] bf16
                           const bf16_t* __restrict__ Wg,   // [E][D][I] bf16
                           const bf16_t* __restrict__ Wu,   // [E][D][I] bf16
                           bf16_t* __restrict__ H,          // [slots][I] bf16
                           const int* __restrict__ tokList, // slot -> token (or null)
                           const int* __restrict__ slotBase,// per-expert base (or null)
                           const int* __restrict__ counts)  // per-expert rows (or null)
{
  int e = blockIdx.z;
  int count = counts ? counts[e] : TOKENS;
  int tileM = blockIdx.x;
  if (tileM * TM >= count) return;
  int base = slotBase ? slotBase[e] : 0;
  int n0 = blockIdx.y * TN;
  const bf16_t* wg = Wg + (size_t)e * DMODEL * DFFN;
  const bf16_t* wu = Wu + (size_t)e * DMODEL * DFFN;

  __shared__ __align__(16) bf16_t As [TM][TKK];
  __shared__ __align__(16) bf16_t Bgt[TN][TKK];
  __shared__ __align__(16) bf16_t But[TN][TKK];

  int tid  = threadIdx.x;
  int lane = tid & 31;
  int wave = tid >> 5;        // 0..7
  int mw   = wave & 3;        // M subtile (16 rows each)
  int nh   = wave >> 2;       // N half (32 cols each)
  int half = lane >> 4;
  int ml   = lane & 15;

  v8f accg[2] = {}; v8f accu[2] = {};

  for (int k0 = 0; k0 < DMODEL; k0 += TKK) {
    // A tile: 64 rows x 32 K, gathered token rows, 8B units (async global->LDS)
    for (int u = tid; u < TM * 8; u += 256) {
      int r = u >> 3, c = (u & 7) * 4;
      int row = tileM * TM + r;
      if (row < count) {
        int t = tokList ? tokList[base + row] : row;
        const bf16_t* src = X + (size_t)t * DMODEL + k0 + c;
#if ASYNC_LDS
        async_copy_b64(&As[r][c], src);
#else
        *(uint2*)&As[r][c] = *(const uint2*)src;
#endif
      } else {
        uint2 z; z.x = 0u; z.y = 0u;
        *(uint2*)&As[r][c] = z;
      }
    }
    // B tiles: rows k0..k0+31, cols n0..n0+63, stored transposed Bt[n][k]
    for (int u = tid; u < TKK * 16; u += 256) {
      int k = u >> 4, c = (u & 15) * 4;
      const bf16_t* sg_ = wg + (size_t)(k0 + k) * DFFN + n0 + c;
      const bf16_t* su_ = wu + (size_t)(k0 + k) * DFFN + n0 + c;
      uint2 g2 = *(const uint2*)sg_;
      uint2 u2 = *(const uint2*)su_;
      if (k0 + TKK < DMODEL) {             // prefetch next K-step (global_prefetch_b8)
        __builtin_prefetch(sg_ + (size_t)TKK * DFFN, 0, 1);
        __builtin_prefetch(su_ + (size_t)TKK * DFFN, 0, 1);
      }
      bf16_t gv[4]; *(uint2*)gv = g2;
      bf16_t uv[4]; *(uint2*)uv = u2;
#pragma unroll
      for (int j = 0; j < 4; ++j) { Bgt[c + j][k] = gv[j]; But[c + j][k] = uv[j]; }
    }
#if ASYNC_LDS
    wait_async_all();
#endif
    __syncthreads();

    v16bf a = load_a_frag(&As[mw * 16 + ml][0], half);
#pragma unroll
    for (int s = 0; s < 2; ++s) {
      int n = nh * 32 + s * 16 + ml;
      v16bf bg = load_b_frag(&Bgt[n][0], half);
      v16bf bu = load_b_frag(&But[n][0], half);
      accg[s] = __builtin_amdgcn_wmma_f32_16x16x32_bf16(
          false, a, false, bg, (short)0, accg[s], false, false);
      accu[s] = __builtin_amdgcn_wmma_f32_16x16x32_bf16(
          false, a, false, bu, (short)0, accu[s], false, false);
    }
    __syncthreads();
  }

  // Epilogue: h = silu(g) * u -> bf16 H at slot rows
#pragma unroll
  for (int s = 0; s < 2; ++s) {
#pragma unroll
    for (int r = 0; r < 8; ++r) {
      int m = r + half * 8;
      int row = tileM * TM + mw * 16 + m;
      if (row < count) {
        float g = accg[s][r];
        float u = accu[s][r];
        float h = g * sigmoidf_(g) * u;
        int n = n0 + nh * 32 + s * 16 + ml;
        H[(size_t)(base + row) * DFFN + n] = (bf16_t)h;
      }
    }
  }
}

// ---------------- Down GEMM + weighted scatter-add ----------------
__launch_bounds__(256)
__global__ void moe_down(const bf16_t* __restrict__ H,     // [slots][I] bf16
                         const bf16_t* __restrict__ Wd,    // [E][I][D] bf16
                         float* __restrict__ out,          // [T][D] f32 (pre-zeroed)
                         const int* __restrict__ tokList,  // slot -> token (or null)
                         const int* __restrict__ slotBase,
                         const int* __restrict__ counts,
                         const float* __restrict__ wSlot)  // per-slot scale
{
  int e = blockIdx.z;
  int count = counts ? counts[e] : TOKENS;
  int tileM = blockIdx.x;
  if (tileM * TM >= count) return;
  int base = slotBase ? slotBase[e] : 0;
  int n0 = blockIdx.y * TN;
  const bf16_t* wd = Wd + (size_t)e * DFFN * DMODEL;

  __shared__ __align__(16) bf16_t As[TM][TKK];
  __shared__ __align__(16) bf16_t Bt[TN][TKK];

  int tid  = threadIdx.x;
  int lane = tid & 31;
  int wave = tid >> 5;
  int mw   = wave & 3;
  int nh   = wave >> 2;
  int half = lane >> 4;
  int ml   = lane & 15;

  v8f acc[2] = {};

  for (int k0 = 0; k0 < DFFN; k0 += TKK) {
    for (int u = tid; u < TM * 8; u += 256) {
      int r = u >> 3, c = (u & 7) * 4;
      int row = tileM * TM + r;
      if (row < count) {
        const bf16_t* src = H + (size_t)(base + row) * DFFN + k0 + c;
#if ASYNC_LDS
        async_copy_b64(&As[r][c], src);
#else
        *(uint2*)&As[r][c] = *(const uint2*)src;
#endif
      } else {
        uint2 z; z.x = 0u; z.y = 0u;
        *(uint2*)&As[r][c] = z;
      }
    }
    for (int u = tid; u < TKK * 16; u += 256) {
      int k = u >> 4, c = (u & 15) * 4;
      const bf16_t* sw_ = wd + (size_t)(k0 + k) * DMODEL + n0 + c;
      uint2 w2 = *(const uint2*)sw_;
      if (k0 + TKK < DFFN) __builtin_prefetch(sw_ + (size_t)TKK * DMODEL, 0, 1);
      bf16_t wv[4]; *(uint2*)wv = w2;
#pragma unroll
      for (int j = 0; j < 4; ++j) Bt[c + j][k] = wv[j];
    }
#if ASYNC_LDS
    wait_async_all();
#endif
    __syncthreads();

    v16bf a = load_a_frag(&As[mw * 16 + ml][0], half);
#pragma unroll
    for (int s = 0; s < 2; ++s) {
      int n = nh * 32 + s * 16 + ml;
      v16bf b = load_b_frag(&Bt[n][0], half);
      acc[s] = __builtin_amdgcn_wmma_f32_16x16x32_bf16(
          false, a, false, b, (short)0, acc[s], false, false);
    }
    __syncthreads();
  }

#pragma unroll
  for (int s = 0; s < 2; ++s) {
#pragma unroll
    for (int r = 0; r < 8; ++r) {
      int m = r + half * 8;
      int row = tileM * TM + mw * 16 + m;
      if (row < count) {
        int slot = base + row;
        int t = tokList ? tokList[slot] : slot;
        float wv = wSlot[slot];
        int n = n0 + nh * 32 + s * 16 + ml;
        atomicAdd(&out[(size_t)t * DMODEL + n], wv * acc[s][r]);
      }
    }
  }
}

// ---------------- Host-side orchestration ----------------
extern "C" void kernel_launch(void* const* d_in, const int* in_sizes, int n_in,
                              void* d_out, int out_size, void* d_ws, size_t ws_size,
                              hipStream_t stream) {
  (void)in_sizes; (void)n_in; (void)out_size; (void)ws_size;
  const float* x        = (const float*)d_in[0]; // [4,2048,1024]
  const float* gw       = (const float*)d_in[1]; // [1024,8]
  const float* egw      = (const float*)d_in[2]; // [8,1024,2048]
  const float* euw      = (const float*)d_in[3];
  const float* edw      = (const float*)d_in[4]; // [8,2048,1024]
  const float* sgw_proj = (const float*)d_in[5]; // [1024,2048]
  const float* suw      = (const float*)d_in[6];
  const float* sdw      = (const float*)d_in[7]; // [2048,1024]
  const float* segw     = (const float*)d_in[8]; // [1024,1]

  char* ws = (char*)d_ws;
  size_t off = 0;
  auto alloc = [&](size_t b) -> char* {
    char* p = ws + off;
    off = (off + b + 255) & ~(size_t)255;
    return p;
  };
  bf16_t* xb   = (bf16_t*)alloc((size_t)TOKENS * DMODEL * 2);
  bf16_t* wgb  = (bf16_t*)alloc((size_t)NEXPERT * DMODEL * DFFN * 2);
  bf16_t* wub  = (bf16_t*)alloc((size_t)NEXPERT * DMODEL * DFFN * 2);
  bf16_t* wdb  = (bf16_t*)alloc((size_t)NEXPERT * DFFN * DMODEL * 2);
  bf16_t* sgb  = (bf16_t*)alloc((size_t)DMODEL * DFFN * 2);
  bf16_t* sub_ = (bf16_t*)alloc((size_t)DMODEL * DFFN * 2);
  bf16_t* sdb  = (bf16_t*)alloc((size_t)DFFN * DMODEL * 2);
  bf16_t* He   = (bf16_t*)alloc((size_t)NSLOTS * DFFN * 2);
  bf16_t* Hs   = (bf16_t*)alloc((size_t)TOKENS * DFFN * 2);
  int*   topi     = (int*)alloc((size_t)TOKENS * 2 * 4);
  float* topw     = (float*)alloc((size_t)TOKENS * 2 * 4);
  float* sgate    = (float*)alloc((size_t)TOKENS * 4);
  int*   counts   = (int*)alloc(NEXPERT * 4);
  int*   slotBase = (int*)alloc((NEXPERT + 1) * 4);
  int*   fill     = (int*)alloc(NEXPERT * 4);
  int*   tokList  = (int*)alloc((size_t)NSLOTS * 4);
  float* tokW     = (float*)alloc((size_t)NSLOTS * 4);

  float* out_final = (float*)d_out;
  float* out_gate  = out_final + (size_t)TOKENS * DMODEL;

  auto cvtLaunch = [&](const float* s, bf16_t* d, size_t n) {
    int blocks = (int)((n / 4 + 255) / 256);
    cvt_f32_bf16<<<blocks, 256, 0, stream>>>(s, d, (int)n);
  };
  cvtLaunch(x,        xb,  (size_t)TOKENS * DMODEL);
  cvtLaunch(egw,      wgb, (size_t)NEXPERT * DMODEL * DFFN);
  cvtLaunch(euw,      wub, (size_t)NEXPERT * DMODEL * DFFN);
  cvtLaunch(edw,      wdb, (size_t)NEXPERT * DFFN * DMODEL);
  cvtLaunch(sgw_proj, sgb, (size_t)DMODEL * DFFN);
  cvtLaunch(suw,      sub_,(size_t)DMODEL * DFFN);
  cvtLaunch(sdw,      sdb, (size_t)DFFN * DMODEL);

  (void)hipMemsetAsync(counts, 0, NEXPERT * sizeof(int), stream);
  zero_f32<<<(TOKENS * DMODEL + 255) / 256, 256, 0, stream>>>(out_final, TOKENS * DMODEL);

  router_kernel<<<TOKENS / 256, 256, 0, stream>>>(x, gw, segw, out_gate,
                                                  topi, topw, sgate, counts);
  prefix_kernel<<<1, 32, 0, stream>>>(counts, slotBase, fill);
  build_lists<<<TOKENS / 256, 256, 0, stream>>>(topi, topw, slotBase, fill, tokList, tokW);

  dim3 blk(256);
  dim3 g1(TOKENS / TM, DFFN / TN, NEXPERT);
  moe_gateup<<<g1, blk, 0, stream>>>(xb, wgb, wub, He, tokList, slotBase, counts);
  dim3 g1s(TOKENS / TM, DFFN / TN, 1);
  moe_gateup<<<g1s, blk, 0, stream>>>(xb, sgb, sub_, Hs, nullptr, nullptr, nullptr);

  dim3 g2(TOKENS / TM, DMODEL / TN, NEXPERT);
  moe_down<<<g2, blk, 0, stream>>>(He, wdb, out_final, tokList, slotBase, counts, tokW);
  dim3 g2s(TOKENS / TM, DMODEL / TN, 1);
  moe_down<<<g2s, blk, 0, stream>>>(Hs, sdb, out_final, nullptr, nullptr, nullptr, sgate);
}
